// QKVAttention_17575006175610
// MI455X (gfx1250) — matrix operand order repaired and checked by reading, not compiled, not run on hardware
//
#include <hip/hip_runtime.h>
#include <hip/hip_bf16.h>

typedef __bf16 bf16_t;
typedef __attribute__((ext_vector_type(16))) __bf16 v16bf;
typedef __attribute__((ext_vector_type(8)))  __bf16 v8bf;
typedef __attribute__((ext_vector_type(8)))  float  v8f;
typedef __attribute__((ext_vector_type(4)))  float  v4f;
typedef __attribute__((ext_vector_type(4)))  unsigned int v4u;
typedef __attribute__((ext_vector_type(8)))  int    v8i;
typedef __attribute__((ext_vector_type(4)))  int    v4i;

#define E_DIM 1024
#define NHEAD 16
#define HDIM  64
#define NTOK  16384   // B*S = 4*4096

// ---------------------------------------------------------------------------
// Prep 0a: x f32 -> bf16 (one-time, removes f32 A traffic + cvt from GEMM loop)
// ---------------------------------------------------------------------------
__global__ __launch_bounds__(256)
void convert_x_bf16(const float* __restrict__ X, bf16_t* __restrict__ Xb)
{
    const size_t idx = ((size_t)blockIdx.x * 256 + threadIdx.x) * 8;
    v4f x0 = *(const v4f*)(X + idx);
    v4f x1 = *(const v4f*)(X + idx + 4);
    v8bf y;
#pragma unroll
    for (int w = 0; w < 4; ++w) { y[w] = (bf16_t)x0[w]; y[4 + w] = (bf16_t)x1[w]; }
    *(v8bf*)(Xb + idx) = y;
}

// ---------------------------------------------------------------------------
// Prep 0b: W[kk][n] f32 -> WT[z][n][kk] bf16 (transposed so the WMMA B operand
// tile is a dense row-major 2D bf16 tile -> TDM-friendly)
// ---------------------------------------------------------------------------
__global__ __launch_bounds__(256)
void transpose_w_bf16(const float* __restrict__ Wq, const float* __restrict__ Wk,
                      const float* __restrict__ Wv, bf16_t* __restrict__ WT)
{
    const int z = blockIdx.z;
    const float* W = (z == 0) ? Wq : (z == 1) ? Wk : Wv;
    __shared__ float tile[32][33];
    const int n0  = blockIdx.x * 32;
    const int kk0 = blockIdx.y * 32;
    const int tid = threadIdx.x;
    const int c = tid & 31, rb = tid >> 5;
#pragma unroll
    for (int i = 0; i < 4; ++i) {
        const int r = rb + i * 8;
        tile[r][c] = W[(size_t)(kk0 + r) * E_DIM + n0 + c];
    }
    __syncthreads();
#pragma unroll
    for (int i = 0; i < 4; ++i) {
        const int r = rb + i * 8;
        WT[((size_t)z << 20) + (size_t)(n0 + r) * E_DIM + kk0 + c] = (bf16_t)tile[c][r];
    }
}

// ---------------------------------------------------------------------------
// Stage 1: Y = Xb @ W + b  (bf16 WMMA, f32 acc), W tiles streamed into LDS by
// the Tensor Data Mover, double-buffered per wave (no barriers in K-loop).
// grid = (NTOK/128, E_DIM/64, 3), block = 256 (8 wave32).
// Each wave: 16(M) x 64(N) strip, K stepped by 32, private 2x4KB LDS buffers.
// ---------------------------------------------------------------------------
__global__ __launch_bounds__(256)
void qkv_gemm_tdm(const bf16_t* __restrict__ Xb,
                  const bf16_t* __restrict__ WT,   // [3][1024 n][1024 kk] bf16
                  const float* __restrict__ bq, const float* __restrict__ bk,
                  const float* __restrict__ bv,
                  bf16_t* __restrict__ Qo, bf16_t* __restrict__ Ko,
                  bf16_t* __restrict__ VTo)
{
    const int z = blockIdx.z;
    const float* bm = (z == 0) ? bq : (z == 1) ? bk : bv;
    bf16_t*      Y  = (z == 0) ? Qo : (z == 1) ? Ko : VTo;

    const int t0   = blockIdx.x * 128;
    const int n0   = blockIdx.y * 64;
    const int tid  = threadIdx.x;
    const int wave = tid >> 5;
    const int lane = tid & 31;
    const int l16  = lane & 15;
    const int g    = lane >> 4;
    const int wuni = __builtin_amdgcn_readfirstlane(wave);  // SGPR wave id

    // per-wave double-buffered B tile: [wave][buf][64 n][32 kk] bf16
    __shared__ __align__(128) bf16_t lds[8][2][64 * 32];

    // ---- TDM descriptor (D#) ----
    // group1: data_size=2B(code 1); tensor_dim0=1024, tensor_dim1=1024;
    //         tile_dim0=32, tile_dim1=64; tensor_dim0_stride=1024
    const v8i g1 = { 0x00010000, 0x04000000, 0x04000000, 0x00200000,
                     64, 1024, 0, 0 };
    const v4i gz4 = { 0, 0, 0, 0 };
    const v8i gz8 = { 0, 0, 0, 0, 0, 0, 0, 0 };
    const unsigned long long wt_base =
        (unsigned long long)(uintptr_t)(WT + ((size_t)z << 20) + (size_t)n0 * E_DIM);

    auto tdm_issue = [&](int buf, int kk0) {
        const unsigned int  ldsa = (unsigned int)(uintptr_t)(&lds[wuni][buf][0]);
        const unsigned long long ga = wt_base + (unsigned long long)kk0 * 2;
        v4u g0;
        g0[0] = 1u;                                   // count=1, user mode
        g0[1] = ldsa;                                 // lds_addr
        g0[2] = (unsigned int)ga;                     // global_addr[31:0]
        g0[3] = (unsigned int)((ga >> 32) & 0x1FFFFFFu) | (2u << 30); // addr hi | type=2
        __builtin_amdgcn_tensor_load_to_lds(g0, g1, gz4, gz4, gz8, 0);
    };

    v8f acc[4] = {};
    const bf16_t* xrow = Xb + (size_t)(t0 + wave * 16 + l16) * E_DIM;

    auto compute = [&](int i) {
        const int kk0 = i * 32;
        // A operand: 16x32 bf16 slice of Xb (contiguous layout K = 16g + w + 8h)
        v16bf a;
        v8bf alo = *(const v8bf*)(xrow + kk0 + 16 * g);
        v8bf ahi = *(const v8bf*)(xrow + kk0 + 16 * g + 8);
#pragma unroll
        for (int w = 0; w < 8; ++w) { a[w] = alo[w]; a[8 + w] = ahi[w]; }
        const bf16_t* bt = &lds[wave][i & 1][0];
#pragma unroll
        for (int nt = 0; nt < 4; ++nt) {
            v16bf b = *(const v16bf*)(bt + (nt * 16 + l16) * 32 + 16 * g);
            acc[nt] = __builtin_amdgcn_wmma_f32_16x16x32_bf16(
                false, a, false, b, (short)0, acc[nt], false, false);
        }
    };

    tdm_issue(0, 0);
    for (int i = 0; i < 31; ++i) {
        tdm_issue((i + 1) & 1, (i + 1) * 32);        // prefetch next tile
        __builtin_amdgcn_s_wait_tensorcnt(1);        // oldest (tile i) landed
        asm volatile("" ::: "memory");
        compute(i);
        asm volatile("" ::: "memory");
    }
    __builtin_amdgcn_s_wait_tensorcnt(0);
    asm volatile("" ::: "memory");
    compute(31);

    // epilogue: bias add in f32, store bf16 (V stored per-token transposed)
    const int trow = t0 + wave * 16;
#pragma unroll
    for (int nt = 0; nt < 4; ++nt) {
        const int   e  = n0 + nt * 16 + l16;
        const float bb = bm[e];
        const int   eo = (z == 2) ? ((e & 63) * 16 + (e >> 6)) : e;
#pragma unroll
        for (int j = 0; j < 8; ++j) {
            const int t = trow + j + 8 * g;          // C layout: M = j + 8*(lane/16)
            Y[(size_t)t * E_DIM + eo] = (bf16_t)(acc[nt][j] + bb);
        }
    }
}

// ---------------------------------------------------------------------------
// Stage 2: per-token head mixing, one wave per token.
//   att(16x16) = q(16x64) @ k^T  -> 2 chained bf16 WMMAs (K=64)
//   out(16x64) = att @ v         -> 4 bf16 WMMAs, K padded 16->32
// ---------------------------------------------------------------------------
__global__ __launch_bounds__(256)
void attn_head_wmma(const bf16_t* __restrict__ Q, const bf16_t* __restrict__ K,
                    const bf16_t* __restrict__ VT, float* __restrict__ Out)
{
    const int tid  = threadIdx.x;
    const int wave = tid >> 5;
    const int lane = tid & 31;
    const int l16  = lane & 15;
    const int g    = lane >> 4;
    const int t    = blockIdx.x * 8 + wave;

    const bf16_t* q  = Q  + (size_t)t * E_DIM;   // [h][d]
    const bf16_t* k  = K  + (size_t)t * E_DIM;   // [h][d]
    const bf16_t* vt = VT + (size_t)t * E_DIM;   // [d][h] (pre-transposed)

    __shared__ __align__(16) float lds_att[8][16][16];

    v8f att = {};
#pragma unroll
    for (int d0 = 0; d0 < HDIM; d0 += 32) {
        v16bf a, b;
        v8bf alo = *(const v8bf*)(q + l16 * HDIM + d0 + 16 * g);
        v8bf ahi = *(const v8bf*)(q + l16 * HDIM + d0 + 16 * g + 8);
#pragma unroll
        for (int w = 0; w < 8; ++w) { a[w] = alo[w]; a[8 + w] = ahi[w]; }
        b = *(const v16bf*)(k + l16 * HDIM + d0 + 16 * g);
        att = __builtin_amdgcn_wmma_f32_16x16x32_bf16(
            false, a, false, b, (short)0, att, false, false);
    }

#pragma unroll
    for (int j = 0; j < 8; ++j)
        lds_att[wave][j + 8 * g][l16] = att[j];
    __syncthreads();

    v16bf a2;
#pragma unroll
    for (int w = 0; w < 8; ++w) {
        a2[w]     = (g == 0) ? (bf16_t)lds_att[wave][l16][w] : (bf16_t)0.0f;      // K = w (g==0)
        a2[8 + w] = (g == 0) ? (bf16_t)lds_att[wave][l16][8 + w] : (bf16_t)0.0f;  // K = 8 + w (g==0); g==1 zero padding
    }

    const v8f czero = {};
#pragma unroll
    for (int nt = 0; nt < 4; ++nt) {
        v16bf b2 = {};
        if (g == 0)
            b2 = *(const v16bf*)(vt + (nt * 16 + l16) * 16);
        v8f o = __builtin_amdgcn_wmma_f32_16x16x32_bf16(
            false, a2, false, b2, (short)0, czero, false, false);
#pragma unroll
        for (int j = 0; j < 8; ++j) {
            const int h = j + 8 * g;
            Out[(size_t)t * E_DIM + h * HDIM + nt * 16 + l16] = o[j];
        }
    }
}

// ---------------------------------------------------------------------------
extern "C" void kernel_launch(void* const* d_in, const int* in_sizes, int n_in,
                              void* d_out, int out_size, void* d_ws, size_t ws_size,
                              hipStream_t stream) {
    const float* x  = (const float*)d_in[0];
    const float* Wq = (const float*)d_in[1];
    const float* bq = (const float*)d_in[2];
    const float* Wk = (const float*)d_in[3];
    const float* bk = (const float*)d_in[4];
    const float* Wv = (const float*)d_in[5];
    const float* bv = (const float*)d_in[6];

    // workspace layout (bf16): Q, K, VT (32MB each), Xb (32MB), WT (6MB)
    bf16_t* Qw  = (bf16_t*)d_ws;
    bf16_t* Kw  = Qw  + (size_t)NTOK * E_DIM;
    bf16_t* VTw = Kw  + (size_t)NTOK * E_DIM;
    bf16_t* Xb  = VTw + (size_t)NTOK * E_DIM;
    bf16_t* WTw = Xb  + (size_t)NTOK * E_DIM;

    convert_x_bf16<<<dim3((NTOK * (E_DIM / 8)) / 256), 256, 0, stream>>>(x, Xb);
    transpose_w_bf16<<<dim3(E_DIM / 32, E_DIM / 32, 3), 256, 0, stream>>>(Wq, Wk, Wv, WTw);

    qkv_gemm_tdm<<<dim3(NTOK / 128, E_DIM / 64, 3), 256, 0, stream>>>(
        Xb, WTw, bq, bk, bv, Qw, Kw, VTw);

    attn_head_wmma<<<dim3(NTOK / 8), 256, 0, stream>>>(
        Qw, Kw, VTw, (float*)d_out);
}
